// NonLocalModule_12446815224299
// MI455X (gfx1250) — matrix operand order repaired and checked by reading, not compiled, not run on hardware
//
#include <hip/hip_runtime.h>
#include <math.h>

#define BB 4
#define CC 256
#define NN 4096
#define CLt 32
#define EPSF 1e-5f

typedef __attribute__((ext_vector_type(16))) _Float16 v16h;
typedef __attribute__((ext_vector_type(8)))  _Float16 v8h;
typedef __attribute__((ext_vector_type(8)))  float    v8f;

// DPP row-rotate (16-lane row; stays within each half of the wave32).
template <int CTRL>
__device__ __forceinline__ float dpp_rot(float x) {
  int r = __builtin_amdgcn_update_dpp(0, __float_as_int(x), CTRL, 0xF, 0xF, true);
  return __int_as_float(r);
}
// All-reduce max over the 16-lane row via rotations 1,2,4,8 (pure VALU).
__device__ __forceinline__ float hmax16(float v) {
  v = fmaxf(v, dpp_rot<0x121>(v));   // row_ror:1
  v = fmaxf(v, dpp_rot<0x122>(v));   // row_ror:2
  v = fmaxf(v, dpp_rot<0x124>(v));   // row_ror:4
  v = fmaxf(v, dpp_rot<0x128>(v));   // row_ror:8
  return v;
}

// ---------------------------------------------------------------------------
// Fold eval-mode BatchNorm into weights/biases.
// ---------------------------------------------------------------------------
__global__ void nl_prep(const float* w1, const float* w2, const float* w3,
                        const float* w_out,
                        const float* s1, const float* b1, const float* m1, const float* v1,
                        const float* s2, const float* b2, const float* m2, const float* v2,
                        const float* s3, const float* b3, const float* m3, const float* v3,
                        const float* s4, const float* b4, const float* m4, const float* v4,
                        float* Wf, float* Bf, float* Wo, float* Bo) {
  int t = blockIdx.x * blockDim.x + threadIdx.x;
  const float* Ws[3]  = {w1, w2, w3};
  const float* Ss[3]  = {s1, s2, s3};
  const float* Bs[3]  = {b1, b2, b3};
  const float* Ms[3]  = {m1, m2, m3};
  const float* Vrs[3] = {v1, v2, v3};
  if (t < 3 * CLt * CC) {
    int which = t / (CLt * CC);
    int rem = t - which * (CLt * CC);
    int o = rem / CC;
    float inv = Ss[which][o] * rsqrtf(Vrs[which][o] + EPSF);
    Wf[t] = Ws[which][rem] * inv;
  }
  if (t < 3 * CLt) {
    int which = t / CLt, o = t - which * CLt;
    float inv = Ss[which][o] * rsqrtf(Vrs[which][o] + EPSF);
    Bf[t] = Bs[which][o] - Ms[which][o] * inv;
  }
  if (t < CC * CLt) {
    int o = t / CLt;
    float inv = s4[o] * rsqrtf(v4[o] + EPSF);
    Wo[t] = w_out[t] * inv;
  }
  if (t < CC) {
    float inv = s4[t] * rsqrtf(v4[t] + EPSF);
    Bo[t] = b4[t] - m4[t] * inv;
  }
}

// ---------------------------------------------------------------------------
// QKV projection: qkv[z][b][n][o] = f16( relu( Wf[z] @ x[b,:,n] + Bf[z] ) )
// ---------------------------------------------------------------------------
__global__ __launch_bounds__(256) void nl_qkv(const float* __restrict__ x,
                                              const float* __restrict__ Wf,
                                              const float* __restrict__ Bf,
                                              _Float16* __restrict__ qkv) {
  int z = blockIdx.z, b = blockIdx.y;
  int n = blockIdx.x * 256 + threadIdx.x;
  const float* W  = Wf + (size_t)z * (CLt * CC);
  const float* Bb = Bf + z * CLt;
  float acc[CLt];
#pragma unroll
  for (int o = 0; o < CLt; ++o) acc[o] = Bb[o];
  const float* xp = x + (size_t)b * CC * NN + n;
  for (int c = 0; c < CC; ++c) {
    float xv = xp[(size_t)c * NN];
#pragma unroll
    for (int o = 0; o < CLt; ++o) acc[o] = fmaf(W[o * CC + c], xv, acc[o]);
  }
  _Float16* dst = qkv + ((size_t)z * BB * NN + (size_t)b * NN + n) * CLt;
#pragma unroll
  for (int g = 0; g < 4; ++g) {
    v8h pk;
#pragma unroll
    for (int j = 0; j < 8; ++j) pk[j] = (_Float16)fmaxf(acc[g * 8 + j], 0.0f);
    *(v8h*)(dst + g * 8) = pk;
  }
}

// ---------------------------------------------------------------------------
// Flash attention. 8 waves/WG, wave owns 16 queries, WG shares 64-key K/V
// tiles in LDS. K tile filled with global_load_async_to_lds_b128 (ASYNCcnt),
// row-max via DPP rotations, row-sum via an extra P x ones WMMA whose D-
// layout matches the O accumulator (denominator for free, no shuffles).
// ---------------------------------------------------------------------------
__global__ __launch_bounds__(256) void nl_attn(const _Float16* __restrict__ q,
                                               const _Float16* __restrict__ k,
                                               const _Float16* __restrict__ v,
                                               float* __restrict__ O) {
  __shared__ _Float16 Ks[64 * 32];        // [key][dim]
  __shared__ _Float16 Vs[32 * 64];        // [dim][key] (transposed)
  __shared__ _Float16 Ps[8 * 16 * 64];    // per-wave P scratch [16 q][64 key]

  const int tid  = threadIdx.x;
  const int lane = tid & 31, wave = tid >> 5;
  const int b    = blockIdx.y;
  const int qrow = blockIdx.x * 128 + wave * 16;
  const int mrow = lane & 15, hi = lane >> 4;

  // Q A-fragment (16x32 f16): K=e+16*hi
  const _Float16* qp = q + ((size_t)b * NN + qrow + mrow) * CLt;
  v16h aQ;
  {
    v8h lo = *(const v8h*)(qp + hi * 16);
    v8h h8 = *(const v8h*)(qp + hi * 16 + 8);
#pragma unroll
    for (int i = 0; i < 8; ++i) { aQ[i] = lo[i]; aQ[8 + i] = h8[i]; }
  }

  // Constant ones B operand (32x16 of 1.0) for the row-sum WMMA.
  v16h bOnes;
#pragma unroll
  for (int i = 0; i < 16; ++i) bOnes[i] = (_Float16)1.0f;

  v8f acc0 = {}, acc1 = {}, acc2 = {};   // O dims 0-15, 16-31, row-sum l
  float mrun[8];
#pragma unroll
  for (int r = 0; r < 8; ++r) mrun[r] = -INFINITY;
  _Float16* Pw = Ps + wave * (16 * 64);

  const int key = tid >> 2, seg = tid & 3;

  for (int kt = 0; kt < NN; kt += 64) {
    __syncthreads();
    {
      // K tile: async global->LDS copy (16B per thread), tracked by ASYNCcnt.
      unsigned lds_dst = (unsigned)(uintptr_t)(Ks + key * 32 + seg * 8);
      unsigned long long gsrc =
          (unsigned long long)(uintptr_t)(k + ((size_t)b * NN + kt + key) * CLt + seg * 8);
      asm volatile("global_load_async_to_lds_b128 %0, %1, off"
                   :: "v"(lds_dst), "v"(gsrc) : "memory");
      // V tile: load + transpose through VGPRs.
      const _Float16* vp = v + ((size_t)b * NN + kt + key) * CLt + seg * 8;
      v8h vv = *(const v8h*)vp;
#pragma unroll
      for (int j = 0; j < 8; ++j) Vs[(seg * 8 + j) * 64 + key] = vv[j];
      // Prefetch next tile into cache.
      int nkt = (kt + 64 < NN) ? kt + 64 : 0;
      __builtin_prefetch((const void*)(k + ((size_t)b * NN + nkt + key) * CLt + seg * 8), 0, 1);
      __builtin_prefetch((const void*)(v + ((size_t)b * NN + nkt + key) * CLt + seg * 8), 0, 1);
    }
    asm volatile("s_wait_asynccnt 0x0" ::: "memory");
    __syncthreads();

    // S tiles: 16 queries x 64 keys as four 16x16 WMMAs
    v8f S[4];
#pragma unroll
    for (int nt = 0; nt < 4; ++nt) {
      const _Float16* kp = Ks + (nt * 16 + mrow) * 32 + hi * 16;  // B: K=e+16*hi
      v8h lo = *(const v8h*)kp;
      v8h h8 = *(const v8h*)(kp + 8);
      v16h bK;
#pragma unroll
      for (int i = 0; i < 8; ++i) { bK[i] = lo[i]; bK[8 + i] = h8[i]; }
      v8f zero = {};
      S[nt] = __builtin_amdgcn_wmma_f32_16x16x32_f16(false, aQ, false, bK,
                                                     (short)0, zero, false, false);
    }

    // Online softmax: row-max via DPP, exponentiate, stage P to LDS as f16.
    float scl[8];
#pragma unroll
    for (int r = 0; r < 8; ++r) {
      float mx = fmaxf(fmaxf(S[0][r], S[1][r]), fmaxf(S[2][r], S[3][r]));
      mx = hmax16(mx);
      float mnew = fmaxf(mrun[r], mx);
      scl[r] = __expf(mrun[r] - mnew);
      mrun[r] = mnew;
#pragma unroll
      for (int nt = 0; nt < 4; ++nt) S[nt][r] = __expf(S[nt][r] - mnew);
    }
#pragma unroll
    for (int r = 0; r < 8; ++r) {
      acc0[r] *= scl[r];
      acc1[r] *= scl[r];
      acc2[r] *= scl[r];
    }
#pragma unroll
    for (int nt = 0; nt < 4; ++nt)
#pragma unroll
      for (int r = 0; r < 8; ++r)
        Pw[(r + 8 * hi) * 64 + nt * 16 + mrow] = (_Float16)S[nt][r];

    // O += P.V and l += P.1 : two K-chunks of 32 keys.
#pragma unroll
    for (int kc = 0; kc < 2; ++kc) {
      const _Float16* pp = Pw + mrow * 64 + kc * 32 + hi * 16;
      v8h lo = *(const v8h*)pp;
      v8h h8 = *(const v8h*)(pp + 8);
      v16h aP;
#pragma unroll
      for (int i = 0; i < 8; ++i) { aP[i] = lo[i]; aP[8 + i] = h8[i]; }
      {
        const _Float16* vp0 = Vs + mrow * 64 + kc * 32 + hi * 16;
        v8h b0 = *(const v8h*)vp0;
        v8h b1 = *(const v8h*)(vp0 + 8);
        v16h bV;
#pragma unroll
        for (int i = 0; i < 8; ++i) { bV[i] = b0[i]; bV[8 + i] = b1[i]; }
        acc0 = __builtin_amdgcn_wmma_f32_16x16x32_f16(false, aP, false, bV,
                                                      (short)0, acc0, false, false);
      }
      {
        const _Float16* vp1 = Vs + (16 + mrow) * 64 + kc * 32 + hi * 16;
        v8h b0 = *(const v8h*)vp1;
        v8h b1 = *(const v8h*)(vp1 + 8);
        v16h bV;
#pragma unroll
        for (int i = 0; i < 8; ++i) { bV[i] = b0[i]; bV[8 + i] = b1[i]; }
        acc1 = __builtin_amdgcn_wmma_f32_16x16x32_f16(false, aP, false, bV,
                                                      (short)0, acc1, false, false);
      }
      // row-sum: every column of P x ones equals sum_k P[m][k]
      acc2 = __builtin_amdgcn_wmma_f32_16x16x32_f16(false, aP, false, bOnes,
                                                    (short)0, acc2, false, false);
    }
  }

  // finalize: O[b][n][d] = acc / l   (l sits in acc2, same D-layout)
#pragma unroll
  for (int r = 0; r < 8; ++r) {
    float invl = 1.0f / acc2[r];
    size_t row = (size_t)b * NN + qrow + r + 8 * hi;
    O[row * CLt + mrow]      = acc0[r] * invl;
    O[row * CLt + 16 + mrow] = acc1[r] * invl;
  }
}

// ---------------------------------------------------------------------------
// Output projection + BN + ReLU + residual: out = gamma*relu(Wo@O + Bo) + x
// ---------------------------------------------------------------------------
__global__ __launch_bounds__(256) void nl_out(const float* __restrict__ x,
                                              const float* __restrict__ O,
                                              const float* __restrict__ Wo,
                                              const float* __restrict__ Bo,
                                              const float* __restrict__ gp,
                                              float* __restrict__ out) {
  __shared__ float Osh[256 * 33];   // stride 33: conflict-free column reads
  int t = threadIdx.x, b = blockIdx.y, nb = blockIdx.x * 256;
  const float* orow = O + ((size_t)b * NN + nb + t) * CLt;
#pragma unroll
  for (int g = 0; g < 8; ++g) {
    float4 f = *(const float4*)(orow + g * 4);
    Osh[t * 33 + g * 4 + 0] = f.x;
    Osh[t * 33 + g * 4 + 1] = f.y;
    Osh[t * 33 + g * 4 + 2] = f.z;
    Osh[t * 33 + g * 4 + 3] = f.w;
  }
  __syncthreads();
  float g = gp[0];
  for (int co = 0; co < CC; ++co) {
    const float* wr = Wo + co * CLt;   // uniform -> scalar loads
    float acc = Bo[co];
#pragma unroll
    for (int cl = 0; cl < CLt; ++cl) acc = fmaf(wr[cl], Osh[t * 33 + cl], acc);
    size_t oi = ((size_t)b * CC + co) * NN + nb + t;
    out[oi] = g * fmaxf(acc, 0.0f) + x[oi];
  }
}

// ---------------------------------------------------------------------------
extern "C" void kernel_launch(void* const* d_in, const int* in_sizes, int n_in,
                              void* d_out, int out_size, void* d_ws, size_t ws_size,
                              hipStream_t stream) {
  const float* x     = (const float*)d_in[0];
  const float* w1    = (const float*)d_in[1];
  const float* w2    = (const float*)d_in[2];
  const float* w3    = (const float*)d_in[3];
  const float* w_out = (const float*)d_in[4];
  const float* s1 = (const float*)d_in[5],  *b1 = (const float*)d_in[6];
  const float* m1 = (const float*)d_in[7],  *v1 = (const float*)d_in[8];
  const float* s2 = (const float*)d_in[9],  *b2 = (const float*)d_in[10];
  const float* m2 = (const float*)d_in[11], *v2 = (const float*)d_in[12];
  const float* s3 = (const float*)d_in[13], *b3 = (const float*)d_in[14];
  const float* m3 = (const float*)d_in[15], *v3 = (const float*)d_in[16];
  const float* s4 = (const float*)d_in[17], *b4 = (const float*)d_in[18];
  const float* m4 = (const float*)d_in[19], *v4 = (const float*)d_in[20];
  const float* gamma = (const float*)d_in[21];

  float* ws = (float*)d_ws;
  float* Wf = ws;                    // 3*32*256 = 24576 f32
  float* Bf = ws + 24576;            // 96
  float* Wo = ws + 24672;            // 8192
  float* Bo = ws + 32864;            // 256
  _Float16* qkv = (_Float16*)(ws + 33120);          // 3 * B*N*32 f16
  float* Obuf = ws + 33120 + 786432;                // B*N*32 f32

  float* out = (float*)d_out;
  const size_t QKV1 = (size_t)BB * NN * CLt;

  nl_prep<<<96, 256, 0, stream>>>(w1, w2, w3, w_out,
                                  s1, b1, m1, v1, s2, b2, m2, v2,
                                  s3, b3, m3, v3, s4, b4, m4, v4,
                                  Wf, Bf, Wo, Bo);
  nl_qkv<<<dim3(NN / 256, BB, 3), 256, 0, stream>>>(x, Wf, Bf, qkv);
  nl_attn<<<dim3(NN / 128, BB), 256, 0, stream>>>(qkv, qkv + QKV1, qkv + 2 * QKV1, Obuf);
  nl_out<<<dim3(NN / 256, BB), 256, 0, stream>>>(x, Obuf, Wo, Bo, gamma, out);
}
